// GNNFraudDetector_40733469835427
// MI455X (gfx1250) — matrix operand order repaired and checked by reading, not compiled, not run on hardware
//
#include <hip/hip_runtime.h>

// ---------------------------------------------------------------------------
// GNN fraud detector (4-layer GAT) for gfx1250 (MI455X).
// Dense node-feature transforms run on bf16 WMMA (v_wmma_f32_16x16x32_bf16,
// f32 accumulate). Edge segment-softmax + aggregation use 3 atomic passes
// (fits in 192MB L2). Wave32 everywhere; shuffle reductions for LN/attention.
// ---------------------------------------------------------------------------

#define N_NODES 50000
#define N_EDGES 800000
#define N_TOT   (N_EDGES + N_NODES)   // edges + self loops
#define IN_CH   128
#define HID     256
#define HEADS   8
#define D_HEAD  32
#define OUT_CH  2
#define N_LAYERS 4
#define LN_EPS  1e-5f
#define NEG_SLOPE 0.2f

typedef __attribute__((ext_vector_type(16))) __bf16 v16bf;
typedef __attribute__((ext_vector_type(8)))  float  v8f;
typedef unsigned short ushort_t;
typedef unsigned int   uint_t;

union Frag16 { uint4 u[2]; v16bf v; };

// ---------------------------------------------------------------------------
// f32 -> bf16 (round to nearest even)
// ---------------------------------------------------------------------------
__global__ void f32_to_bf16_kernel(const float* __restrict__ in,
                                   ushort_t* __restrict__ out, int n) {
    int i = blockIdx.x * blockDim.x + threadIdx.x;
    if (i >= n) return;
    uint_t u = __float_as_uint(in[i]);
    uint_t r = (u + 0x7FFFu + ((u >> 16) & 1u)) >> 16;
    out[i] = (ushort_t)r;
}

__global__ void zero_f32_kernel(float* __restrict__ p, int n) {
    int i = blockIdx.x * blockDim.x + threadIdx.x;
    if (i < n) p[i] = 0.0f;
}

// ---------------------------------------------------------------------------
// WMMA GEMM: C[M,N] = A[M,K](bf16,row-major) x B[K,N](bf16,row-major), f32 out.
// Block = 128 threads (4 waves). Each wave: NT 16-col tiles x MT 16-row tiles.
// grid.x = (M/16)/MT, grid.y = N/(64*NT). All fragment loads for one K-chunk
// are issued before any WMMA so the wave takes one bulk s_wait_loadcnt and
// then runs MT*NT WMMAs back-to-back (B and A fragments both reused).
// Fragment layouts per CDNA5 ISA 7.12.2:
//   A (16x32 bf16): lane l -> row (l&15); half g=l>>4; uint4 ap[g], ap[2+g].
//   B (32x16 bf16): lane l -> col (l&15); half g; same K order as A (strided gather).
//   C/D (16x16 f32): VGPR r -> row r + g*8, col = lane&15.
// ---------------------------------------------------------------------------
template<int MT, int NT>
__global__ void gemm_bf16_wmma_kernel(const ushort_t* __restrict__ A,
                                      const ushort_t* __restrict__ B,
                                      float* __restrict__ C,
                                      int K, int N) {
    const int lane    = threadIdx.x & 31;
    const int wave    = threadIdx.x >> 5;
    const int colBase = (blockIdx.y * (blockDim.x >> 5) + wave) * (16 * NT);
    const int mTile0  = blockIdx.x * MT;
    const int g       = lane >> 4;
    const int mr      = lane & 15;

    v8f acc[MT][NT];
#pragma unroll
    for (int t = 0; t < MT; ++t)
#pragma unroll
        for (int n = 0; n < NT; ++n) {
            v8f z = {0.f, 0.f, 0.f, 0.f, 0.f, 0.f, 0.f, 0.f};
            acc[t][n] = z;
        }

    for (int k = 0; k < K; k += 32) {
        // ---- issue ALL fragment loads for this K-chunk first ----
        Frag16 bfrag[NT];
#pragma unroll
        for (int n = 0; n < NT; ++n) {
            const ushort_t* bp =
                B + (size_t)(k + 8 * g) * N + colBase + 16 * n + mr;
            uint_t w[8];
#pragma unroll
            for (int j = 0; j < 8; ++j) {
                const int kk = (j < 4) ? 2 * j : 16 + 2 * (j - 4);
                w[j] = (uint_t)bp[(size_t)kk * N] |
                       ((uint_t)bp[(size_t)(kk + 1) * N] << 16);
            }
            bfrag[n].u[0] = make_uint4(w[0], w[1], w[2], w[3]);
            bfrag[n].u[1] = make_uint4(w[4], w[5], w[6], w[7]);
        }
        Frag16 af[MT];
#pragma unroll
        for (int t = 0; t < MT; ++t) {
            const int row = (mTile0 + t) * 16 + mr;
            const uint4* ap = (const uint4*)(A + (size_t)row * K + k);
            af[t].u[0] = ap[g];
            af[t].u[1] = ap[2 + g];
        }
        // ---- then all WMMAs back-to-back ----
#pragma unroll
        for (int t = 0; t < MT; ++t)
#pragma unroll
            for (int n = 0; n < NT; ++n)
                acc[t][n] = __builtin_amdgcn_wmma_f32_16x16x32_bf16(
                    false, af[t].v, false, bfrag[n].v, (short)0, acc[t][n],
                    false, false);
    }

#pragma unroll
    for (int t = 0; t < MT; ++t) {
        const int rowb = (mTile0 + t) * 16 + g * 8;
#pragma unroll
        for (int n = 0; n < NT; ++n) {
            float* cp = C + (size_t)rowb * N + colBase + 16 * n + mr;
#pragma unroll
            for (int r = 0; r < 8; ++r) cp[(size_t)r * N] = acc[t][n][r];
        }
    }
}

// ---------------------------------------------------------------------------
// Fused bias + LayerNorm + ReLU (+ optional residual add AFTER relu).
// One wave per node; C is 128 or 256 (C/32 channels per lane).
// ---------------------------------------------------------------------------
__global__ void bias_ln_relu_kernel(const float* __restrict__ in,
                                    const float* __restrict__ bias,
                                    const float* __restrict__ gam,
                                    const float* __restrict__ bet,
                                    const float* __restrict__ residual,
                                    float* __restrict__ out,
                                    int C, int nNodes) {
    const int lane = threadIdx.x & 31;
    const int node = blockIdx.x * (blockDim.x >> 5) + (threadIdx.x >> 5);
    if (node >= nNodes) return;
    const int J = C >> 5;
    float vals[8];
    const float* xi = in + (size_t)node * C;
    float s = 0.f;
    for (int j = 0; j < J; ++j) {
        const int c = j * 32 + lane;
        float v = xi[c] + bias[c];
        vals[j] = v;
        s += v;
    }
    for (int o = 16; o; o >>= 1) s += __shfl_xor(s, o, 32);
    const float mean = s / (float)C;
    float vs = 0.f;
    for (int j = 0; j < J; ++j) {
        float d = vals[j] - mean;
        vs += d * d;
    }
    for (int o = 16; o; o >>= 1) vs += __shfl_xor(vs, o, 32);
    const float inv = rsqrtf(vs / (float)C + LN_EPS);
    float* yo = out + (size_t)node * C;
    for (int j = 0; j < J; ++j) {
        const int c = j * 32 + lane;
        float y = gam[c] * (vals[j] - mean) * inv + bet[c];
        y = fmaxf(y, 0.f);
        if (residual) y += residual[(size_t)node * C + c];
        yo[c] = y;
    }
}

// ---------------------------------------------------------------------------
// Per-node attention coefficients: a_src[n,h] = <xh[n,h,:], att_src[h,:]>
// One wave per node; head j handled at lane-parallel dim index = lane.
// ---------------------------------------------------------------------------
__global__ void att_coeff_kernel(const float* __restrict__ xh,
                                 const float* __restrict__ att_src,
                                 const float* __restrict__ att_dst,
                                 float* __restrict__ a_src,
                                 float* __restrict__ a_dst, int nNodes) {
    const int lane = threadIdx.x & 31;
    const int node = blockIdx.x * (blockDim.x >> 5) + (threadIdx.x >> 5);
    if (node >= nNodes) return;
    const float* x = xh + (size_t)node * HID;
    for (int j = 0; j < HEADS; ++j) {
        const float v = x[j * 32 + lane];
        float s = v * att_src[j * 32 + lane];
        float d = v * att_dst[j * 32 + lane];
        for (int o = 16; o; o >>= 1) {
            s += __shfl_xor(s, o, 32);
            d += __shfl_xor(d, o, 32);
        }
        if (lane == 0) {
            a_src[node * HEADS + j] = s;
            a_dst[node * HEADS + j] = d;
        }
    }
}

// order-preserving float <-> uint encoding for atomicMax on floats
__device__ __forceinline__ uint_t fenc(float f) {
    uint_t u = __float_as_uint(f);
    return ((int)u >= 0) ? (u | 0x80000000u) : ~u;
}
__device__ __forceinline__ float fdec(uint_t u) {
    return ((int)u < 0) ? __uint_as_float(u & 0x7FFFFFFFu) : __uint_as_float(~u);
}

__device__ __forceinline__ void edge_sd(const int* __restrict__ ei, int e,
                                        int& s, int& d) {
    if (e < N_EDGES) { s = ei[e]; d = ei[N_EDGES + e]; }
    else             { s = d = e - N_EDGES; }
}

// pass 1: segment max of leaky_relu(a_src[src]+a_dst[dst]) over dst
__global__ void edge_max_kernel(const int* __restrict__ ei,
                                const float* __restrict__ a_src,
                                const float* __restrict__ a_dst,
                                uint_t* __restrict__ segmax) {
    int e = blockIdx.x * blockDim.x + threadIdx.x;
    if (e >= N_TOT) return;
    int s, d; edge_sd(ei, e, s, d);
    for (int j = 0; j < HEADS; ++j) {
        float v = a_src[s * HEADS + j] + a_dst[d * HEADS + j];
        v = v > 0.f ? v : NEG_SLOPE * v;
        atomicMax(&segmax[d * HEADS + j], fenc(v));
    }
}

// pass 2: segment sum of exp(e - max)
__global__ void edge_sum_kernel(const int* __restrict__ ei,
                                const float* __restrict__ a_src,
                                const float* __restrict__ a_dst,
                                const uint_t* __restrict__ segmax,
                                float* __restrict__ segsum) {
    int e = blockIdx.x * blockDim.x + threadIdx.x;
    if (e >= N_TOT) return;
    int s, d; edge_sd(ei, e, s, d);
    for (int j = 0; j < HEADS; ++j) {
        float v = a_src[s * HEADS + j] + a_dst[d * HEADS + j];
        v = v > 0.f ? v : NEG_SLOPE * v;
        float p = __expf(v - fdec(segmax[d * HEADS + j]));
        atomicAdd(&segsum[d * HEADS + j], p);
    }
}

// pass 3: agg[dst] += alpha * xh[src].  One wave per edge; lane covers dim,
// head loop j; alpha computed by lanes 0..7 and broadcast.
__global__ void edge_agg_kernel(const int* __restrict__ ei,
                                const float* __restrict__ a_src,
                                const float* __restrict__ a_dst,
                                const uint_t* __restrict__ segmax,
                                const float* __restrict__ segsum,
                                const float* __restrict__ xh,
                                float* __restrict__ agg) {
    const int lane = threadIdx.x & 31;
    const int e = blockIdx.x * (blockDim.x >> 5) + (threadIdx.x >> 5);
    if (e >= N_TOT) return;
    int s, d; edge_sd(ei, e, s, d);
    float alpha = 0.f;
    if (lane < HEADS) {
        float v = a_src[s * HEADS + lane] + a_dst[d * HEADS + lane];
        v = v > 0.f ? v : NEG_SLOPE * v;
        alpha = __expf(v - fdec(segmax[d * HEADS + lane])) /
                segsum[d * HEADS + lane];
    }
    for (int j = 0; j < HEADS; ++j) {
        const float aj = __shfl(alpha, j, 32);
        const float val = xh[(size_t)s * HID + j * 32 + lane] * aj;
        atomicAdd(&agg[(size_t)d * HID + j * 32 + lane], val);
    }
}

// ---------------------------------------------------------------------------
// Final 128 -> 2 linear (f32, wave per node, shuffle reduce)
// ---------------------------------------------------------------------------
__global__ void final_linear_kernel(const float* __restrict__ h,
                                    const float* __restrict__ w,
                                    const float* __restrict__ b,
                                    float* __restrict__ out, int nNodes) {
    const int lane = threadIdx.x & 31;
    const int node = blockIdx.x * (blockDim.x >> 5) + (threadIdx.x >> 5);
    if (node >= nNodes) return;
    const float* x = h + (size_t)node * (HID / 2);
    float a0 = 0.f, a1 = 0.f;
    for (int j = 0; j < 4; ++j) {
        const int c = j * 32 + lane;
        const float v = x[c];
        a0 += v * w[c * 2 + 0];
        a1 += v * w[c * 2 + 1];
    }
    for (int o = 16; o; o >>= 1) {
        a0 += __shfl_xor(a0, o, 32);
        a1 += __shfl_xor(a1, o, 32);
    }
    if (lane == 0) {
        out[node * 2 + 0] = a0 + b[0];
        out[node * 2 + 1] = a1 + b[1];
    }
}

// ---------------------------------------------------------------------------
// Host-side orchestration
// ---------------------------------------------------------------------------
static inline void cvt(const float* src, ushort_t* dst, int n, hipStream_t s) {
    f32_to_bf16_kernel<<<(n + 255) / 256, 256, 0, s>>>(src, dst, n);
}

extern "C" void kernel_launch(void* const* d_in, const int* in_sizes, int n_in,
                              void* d_out, int out_size, void* d_ws, size_t ws_size,
                              hipStream_t stream) {
    (void)in_sizes; (void)n_in; (void)out_size; (void)ws_size;

    // ---- input indexing: pytree leaves in dict-insertion order ----
    const float* x  = (const float*)d_in[0];                // [50000,128]
    const int*   ei = (const int*)d_in[1];                  // [2,800000]
    const float* in_w    = (const float*)d_in[2];
    const float* in_b    = (const float*)d_in[3];
    const float* in_ln_g = (const float*)d_in[4];
    const float* in_ln_b = (const float*)d_in[5];
    // layers: indices 6 .. 29  (W, att_src, att_dst, bias, ln_g, ln_b) x 4
    const float* c1_w    = (const float*)d_in[30];
    const float* c1_b    = (const float*)d_in[31];
    const float* c1_ln_g = (const float*)d_in[32];
    const float* c1_ln_b = (const float*)d_in[33];
    const float* c2_w    = (const float*)d_in[34];
    const float* c2_b    = (const float*)d_in[35];
    const float* c2_ln_g = (const float*)d_in[36];
    const float* c2_ln_b = (const float*)d_in[37];
    const float* c3_w    = (const float*)d_in[38];
    const float* c3_b    = (const float*)d_in[39];
    float* out = (float*)d_out;

    // ---- workspace layout (all 16B aligned) ----
    char* ws = (char*)d_ws;
    const size_t SZ_HF = (size_t)N_NODES * HID * 4;         // 51.2 MB
    float*    H    = (float*)(ws);                          // node features
    float*    TMP  = (float*)(ws + SZ_HF);                  // gemm out / xh
    float*    AGG  = (float*)(ws + 2 * SZ_HF);              // aggregation
    ushort_t* BF   = (ushort_t*)(ws + 3 * SZ_HF);           // bf16 activations
    char*     p    = ws + 3 * SZ_HF + (size_t)N_NODES * HID * 2;
    float*    ASRC = (float*)(p);  p += (size_t)N_NODES * HEADS * 4;
    float*    ADST = (float*)(p);  p += (size_t)N_NODES * HEADS * 4;
    uint_t*   SMAX = (uint_t*)(p); p += (size_t)N_NODES * HEADS * 4;
    float*    SSUM = (float*)(p);  p += (size_t)N_NODES * HEADS * 4;
    ushort_t* WBF  = (ushort_t*)(p);
    ushort_t* wbf_in = WBF;                                 // 128*256
    ushort_t* wbf_W0 = WBF + IN_CH * HID;                   // 4 x 256*256
    ushort_t* wbf_c1 = wbf_W0 + 4 * HID * HID;              // 256*256
    ushort_t* wbf_c2 = wbf_c1 + HID * HID;                  // 256*128

    // ---- convert weights to bf16 ----
    cvt(in_w, wbf_in, IN_CH * HID, stream);
    for (int l = 0; l < N_LAYERS; ++l)
        cvt((const float*)d_in[6 + 6 * l + 0], wbf_W0 + (size_t)l * HID * HID,
            HID * HID, stream);
    cvt(c1_w, wbf_c1, HID * HID, stream);
    cvt(c2_w, wbf_c2, HID * (HID / 2), stream);

    const dim3 gemmBlk(128);
    const dim3 gemmGrid256(625, 2);   // N=256: (50000/16)/5  x  256/(64*NT=128)
    const dim3 gemmGrid128(625, 1);   // N=128
    const int nodeWaves = (N_NODES + 7) / 8;                // 8 waves / block
    const int edgeThr   = (N_TOT + 255) / 256;
    const int edgeWaves = (N_TOT + 7) / 8;

    // ---- input projection: H = relu(LN(x @ in_w + in_b)) ----
    cvt(x, BF, N_NODES * IN_CH, stream);
    gemm_bf16_wmma_kernel<5, 2><<<gemmGrid256, gemmBlk, 0, stream>>>(
        BF, wbf_in, TMP, IN_CH, HID);
    bias_ln_relu_kernel<<<nodeWaves, 256, 0, stream>>>(
        TMP, in_b, in_ln_g, in_ln_b, nullptr, H, HID, N_NODES);

    // ---- 4 GAT layers ----
    for (int l = 0; l < N_LAYERS; ++l) {
        const float* att_s = (const float*)d_in[6 + 6 * l + 1];
        const float* att_d = (const float*)d_in[6 + 6 * l + 2];
        const float* bias  = (const float*)d_in[6 + 6 * l + 3];
        const float* ln_g  = (const float*)d_in[6 + 6 * l + 4];
        const float* ln_b  = (const float*)d_in[6 + 6 * l + 5];

        // xh = H @ W  (bf16 WMMA, f32 out -> TMP)
        cvt(H, BF, N_NODES * HID, stream);
        gemm_bf16_wmma_kernel<5, 2><<<gemmGrid256, gemmBlk, 0, stream>>>(
            BF, wbf_W0 + (size_t)l * HID * HID, TMP, HID, HID);

        // attention coefficients
        att_coeff_kernel<<<nodeWaves, 256, 0, stream>>>(
            TMP, att_s, att_d, ASRC, ADST, N_NODES);

        // stats init (SMAX 0 sentinel, below enc(-inf); SSUM 0.0f) — contiguous
        zero_f32_kernel<<<(N_NODES * HEADS * 2 + 255) / 256, 256, 0, stream>>>(
            (float*)SMAX, N_NODES * HEADS * 2);
        edge_max_kernel<<<edgeThr, 256, 0, stream>>>(ei, ASRC, ADST, SMAX);
        edge_sum_kernel<<<edgeThr, 256, 0, stream>>>(ei, ASRC, ADST, SMAX, SSUM);

        // weighted aggregation
        zero_f32_kernel<<<(N_NODES * HID + 255) / 256, 256, 0, stream>>>(
            AGG, N_NODES * HID);
        edge_agg_kernel<<<edgeWaves, 256, 0, stream>>>(
            ei, ASRC, ADST, SMAX, SSUM, TMP, AGG);

        // H = relu(LN(AGG + bias)) + H   (in-place residual)
        bias_ln_relu_kernel<<<nodeWaves, 256, 0, stream>>>(
            AGG, bias, ln_g, ln_b, H, H, HID, N_NODES);
    }

    // ---- c1: H = relu(LN(H @ c1_w + c1_b)) ----
    cvt(H, BF, N_NODES * HID, stream);
    gemm_bf16_wmma_kernel<5, 2><<<gemmGrid256, gemmBlk, 0, stream>>>(
        BF, wbf_c1, TMP, HID, HID);
    bias_ln_relu_kernel<<<nodeWaves, 256, 0, stream>>>(
        TMP, c1_b, c1_ln_g, c1_ln_b, nullptr, H, HID, N_NODES);

    // ---- c2: AGG[:, :128] = relu(LN(H @ c2_w + c2_b)) ----
    cvt(H, BF, N_NODES * HID, stream);
    gemm_bf16_wmma_kernel<5, 2><<<gemmGrid128, gemmBlk, 0, stream>>>(
        BF, wbf_c2, TMP, HID, HID / 2);
    bias_ln_relu_kernel<<<nodeWaves, 256, 0, stream>>>(
        TMP, c2_b, c2_ln_g, c2_ln_b, nullptr, AGG, HID / 2, N_NODES);

    // ---- c3: out = AGG @ c3_w + c3_b ----
    final_linear_kernel<<<nodeWaves, 256, 0, stream>>>(
        AGG, c3_w, c3_b, out, N_NODES);
}
